// T5Attention_3453153706487
// MI455X (gfx1250) — matrix operand order repaired and checked by reading, not compiled, not run on hardware
//
#include <hip/hip_runtime.h>
#include <hip/hip_bf16.h>
#include <math.h>

// ---------------------------------------------------------------------------
// T5 attention (bs=2, q=2048, d_model=1024, 16 heads x 64) for gfx1250.
// All GEMMs via v_wmma_f32_16x16x32_f16 (wave32, f32 accumulate).
// K/V tiles staged to LDS with GLOBAL_LOAD_ASYNC_TO_LDS (ASYNCcnt) and
// shared by 4 waves per block; all f16 operands are laid out so every
// fragment load is a contiguous b128 (weights pre-transposed N-major,
// x/context pre-converted to f16, V stored transposed).
// ---------------------------------------------------------------------------

#define DEV __device__ __forceinline__

typedef __attribute__((ext_vector_type(16))) _Float16 v16h;
typedef __attribute__((ext_vector_type(8)))  _Float16 v8h;
typedef __attribute__((ext_vector_type(8)))  float    v8f;
typedef int v4i_t __attribute__((vector_size(16)));     // matches builtin param

constexpr int BS     = 2;
constexpr int SLEN   = 2048;
constexpr int DM     = 1024;
constexpr int H      = 16;
constexpr int DK     = 64;
constexpr int INNER  = H * DK;           // 1024
constexpr int NBUCK  = 64;
constexpr int MAXDST = 50;

// ---------------------- async global->LDS staging --------------------------
#if __has_builtin(__builtin_amdgcn_global_load_async_to_lds_b128)
#define HAVE_ASYNC_LDS 1
#else
#define HAVE_ASYNC_LDS 0
#endif

DEV void stage16B(_Float16* lds_dst, const _Float16* gsrc) {
#if HAVE_ASYNC_LDS
  __builtin_amdgcn_global_load_async_to_lds_b128(
      (__attribute__((address_space(1))) v4i_t*)(v4i_t*)(void*)const_cast<_Float16*>(gsrc),
      (__attribute__((address_space(3))) v4i_t*)(v4i_t*)(void*)lds_dst,
      /*imm offset=*/0, /*cpol=*/0);
#else
  *(uint4*)lds_dst = *(const uint4*)gsrc;
#endif
}

DEV void stage_wait() {
#if HAVE_ASYNC_LDS
#if __has_builtin(__builtin_amdgcn_s_wait_asynccnt)
  __builtin_amdgcn_s_wait_asynccnt(0);
#endif
#endif
}

// ------------------------- WMMA fragment helpers ---------------------------
// Layouts used here (wave32; A/B use contiguous k, not the ISA 7.12.2 table order):
//   A (16x32 f16): lanes 0-15 row M=lane hold K = k0+0..15;
//                  lanes 16-31 row M=lane-16 hold K = k0+16..31.
//   B (32x16 f16): lanes 0-15 hold K=k0+0..15, lanes 16-31 K=k0+16..31, N=lane&15.
//   C/D (16x16 f32, 8 VGPRs): M = r + 8*(lane>=16), N = lane&15.

// A fragment from a row-major f16 matrix: two contiguous 16B vector loads.
DEV v16h load_a_f16(const _Float16* A, int lda, int row0, int k0, int lane) {
  const int half = lane >> 4;
  const int m    = lane & 15;
  const _Float16* rp = A + (size_t)(row0 + m) * lda + k0;
  v8h lo = *(const v8h*)(rp + half * 16);
  v8h hi = *(const v8h*)(rp + half * 16 + 8);
  v16h a;
#pragma unroll
  for (int e = 0; e < 8; ++e) { a[e] = lo[e]; a[8 + e] = hi[e]; }
  return a;
}

// A fragment from a row-major f32 matrix (converts on load).
DEV v16h load_a_f32(const float* A, int lda, int row0, int k0, int lane) {
  const int half = lane >> 4;
  const int m    = lane & 15;
  const float* rp = A + (size_t)(row0 + m) * lda;
  v16h a;
#pragma unroll
  for (int e = 0; e < 8; ++e) a[e]     = (_Float16)rp[k0 + half * 16 + e];
#pragma unroll
  for (int e = 0; e < 8; ++e) a[8 + e] = (_Float16)rp[k0 + half * 16 + 8 + e];
  return a;
}

// B = transpose of a row-major (N x K) f16 matrix M (B[k][n] = M[n][k]);
// per-lane reads are 16 contiguous halfs -> b128 pair.
DEV v16h load_bt_frag(const _Float16* M, int ldm, int col0, int k0, int lane) {
  const int half = lane >> 4;
  const int n    = lane & 15;
  const _Float16* rp = M + (size_t)(col0 + n) * ldm + k0 + half * 16;
  return *(const v16h*)rp;
}

// Same as above but from an LDS tile (row-major (N x Ktile), stride ldm halfs).
DEV v16h load_bt_lds(const _Float16* lds, int ldm, int col0, int k0, int lane) {
  const int half = lane >> 4;
  const int n    = lane & 15;
  return *(const v16h*)(lds + (col0 + n) * ldm + k0 + half * 16);
}

DEV v8f wmma16(v16h a, v16h b, v8f c) {
  return __builtin_amdgcn_wmma_f32_16x16x32_f16(
      /*neg_a=*/false, a, /*neg_b=*/false, b,
      /*c_mod=*/(short)0, c, /*reuse_a=*/false, /*reuse_b=*/false);
}

// ------------------- Kernel 0a: transpose weights to f16 -------------------
// W (DM x INNER) f32 -> Wt (INNER x DM) f16 (N-major). grid: (DM/256, INNER, 4)
__global__ void k_wtrans(const float* __restrict__ Wq, const float* __restrict__ Wk,
                         const float* __restrict__ Wv, const float* __restrict__ Wo,
                         _Float16* __restrict__ wt) {
  const int k = blockIdx.x * 256 + threadIdx.x;   // [0, DM)
  const int n = blockIdx.y;                       // [0, INNER)
  const int m = blockIdx.z;                       // which matrix
  const float* W = (m == 0) ? Wq : (m == 1) ? Wk : (m == 2) ? Wv : Wo;
  _Float16* dst = wt + (size_t)m * INNER * DM;
  dst[(size_t)n * DM + k] = (_Float16)W[(size_t)k * INNER + n];
}

// ---------------------- Kernel 0b: convert x to f16 ------------------------
__global__ void k_xhalf(const float* __restrict__ x, _Float16* __restrict__ xh) {
  const size_t i = (size_t)blockIdx.x * 256 + threadIdx.x;   // < BS*SLEN*DM
  xh[i] = (_Float16)x[i];
}

// ----------------------------- Kernel 1: QKV -------------------------------
// grid: (INNER/16, BS*SLEN/16, 3), block: 32 (one wave -> one 16x16 tile).
// Q,K stored (b,h,s,d); V stored transposed (b,h,d,s).
__global__ void k_qkv(const _Float16* __restrict__ xh,
                      const _Float16* __restrict__ wt,
                      const float* __restrict__ bq, const float* __restrict__ bk,
                      const float* __restrict__ bv,
                      _Float16* __restrict__ qb, _Float16* __restrict__ kb,
                      _Float16* __restrict__ vt) {
  const int lane = threadIdx.x;
  const int tn = blockIdx.x, tm = blockIdx.y, which = blockIdx.z;

  const _Float16* Wt  = wt + (size_t)which * INNER * DM;        // (INNER, DM)
  const float* bias   = (which == 0) ? bq : (which == 1) ? bk : bv;

  const int row0 = tm * 16, col0 = tn * 16;
  v8f acc = {};
#pragma unroll 4
  for (int k0 = 0; k0 < DM; k0 += 32) {
    v16h a = load_a_f16(xh, DM, row0, k0, lane);      // x f16: b128 pair
    v16h b = load_bt_frag(Wt, DM, col0, k0, lane);    // contiguous f16
    acc = wmma16(a, b, acc);
  }
  const int half = lane >> 4, n = lane & 15;
#pragma unroll
  for (int r = 0; r < 8; ++r) {
    const int grow = row0 + r + 8 * half;    // in [0, BS*SLEN)
    const int gcol = col0 + n;               // in [0, INNER)
    const float val = acc[r] + bias[gcol];
    const int b_i = grow / SLEN, s = grow % SLEN;
    const int h   = gcol / DK,   d = gcol % DK;
    const size_t bh = (size_t)b_i * H + h;
    if (which == 2)
      vt[(bh * DK + d) * SLEN + s] = (_Float16)val;             // (b,h,d,s)
    else {
      _Float16* dst = (which == 0) ? qb : kb;
      dst[(bh * SLEN + s) * DK + d] = (_Float16)val;            // (b,h,s,d)
    }
  }
}

// ------------------------ Kernel 2: position bias --------------------------
__global__ void k_pbias(const float* __restrict__ rel_table,
                        float* __restrict__ pb) {
  const size_t idx = (size_t)blockIdx.x * blockDim.x + threadIdx.x;
  const int kk = (int)(idx % SLEN);
  const size_t t = idx / SLEN;
  const int q = (int)(t % SLEN);
  const int h = (int)(t / SLEN);

  const int relp = kk - q;
  const int nb = NBUCK / 2;              // 32
  const int max_exact = nb / 2;          // 16
  const int ret = (relp > 0) ? nb : 0;
  const int n = relp < 0 ? -relp : relp;
  const int nl = n < 1 ? 1 : n;
  int large = max_exact +
      (int)(logf((float)nl / (float)max_exact) *
            (float)(nb - max_exact) /
            logf((float)MAXDST / (float)max_exact));
  if (large > nb - 1) large = nb - 1;
  const int bucket = ret + ((n < max_exact) ? n : large);
  pb[idx] = rel_table[bucket * H + h];
}

// ------------------------- Kernel 3: scores + bias -------------------------
// grid: (SLEN/64, BS*H), block: 128 = 4 waves; each wave owns one q-tile,
// the 4 waves share each K tile staged in LDS via async copy.
__global__ void k_scores(const _Float16* __restrict__ qb,
                         const _Float16* __restrict__ kb,
                         const float* __restrict__ pbias,
                         const int* __restrict__ mask,
                         float* __restrict__ wts) {
  __shared__ alignas(32) _Float16 ldsK[16 * DK];   // 2 KB: K tile (kidx, d)

  const int tid  = threadIdx.x;
  const int lane = tid & 31;
  const int wave = tid >> 5;
  const int qt = blockIdx.x * 4 + wave;
  const int bh = blockIdx.y;
  const int b = bh / H, h = bh % H;

  const _Float16* Q = qb + (size_t)bh * SLEN * DK;   // (SLEN, DK)
  const _Float16* K = kb + (size_t)bh * SLEN * DK;   // (SLEN, DK)
  float* wrow      = wts   + (size_t)bh * SLEN * SLEN + (size_t)qt * 16 * SLEN;
  const float* pb  = pbias + (size_t)h  * SLEN * SLEN + (size_t)qt * 16 * SLEN;
  const int* mrow  = mask + (size_t)b * SLEN;

  // Q fragments stay in registers for the whole k sweep.
  const v16h a0 = load_a_f16(Q, DK, qt * 16, 0,  lane);
  const v16h a1 = load_a_f16(Q, DK, qt * 16, 32, lane);

  const int half = lane >> 4, n = lane & 15;
  // Staging map: 128 threads x 16B = 2 KB tile.
  const int sr = tid >> 3;            // tile row   [0,16)
  const int sc = (tid & 7) * 8;       // half-index [0,64) step 8

  for (int kt = 0; kt < SLEN / 16; ++kt) {
    __syncthreads();   // previous-iteration readers done
    stage16B(&ldsK[sr * DK + sc], K + (size_t)(kt * 16 + sr) * DK + sc);
    stage_wait();
    __syncthreads();   // tile visible to all 4 waves

    if (kt + 1 < SLEN / 16)   // prefetch next K tile (global_prefetch_b8)
      __builtin_prefetch(K + (size_t)(kt + 1) * 16 * DK + tid * 8, 0, 0);

    v16h b0 = load_bt_lds(ldsK, DK, 0, 0,  lane);   // B = K^T
    v16h b1 = load_bt_lds(ldsK, DK, 0, 32, lane);
    v8f acc = {};
    acc = wmma16(a0, b0, acc);
    acc = wmma16(a1, b1, acc);

    const int kcol = kt * 16 + n;
    const float mval = mrow[kcol] ? -__builtin_inff() : 0.0f;
#pragma unroll
    for (int r = 0; r < 8; ++r) {
      const int qr = r + 8 * half;
      wrow[(size_t)qr * SLEN + kcol] = acc[r] + pb[(size_t)qr * SLEN + kcol] + mval;
    }
  }
}

// --------------------------- Kernel 4: softmax -----------------------------
__global__ void k_softmax(float* __restrict__ wts) {
  const size_t row = blockIdx.x;
  float* p = wts + row * (size_t)SLEN;
  __shared__ float red[256];
  const int tid = threadIdx.x;

  float m = -__builtin_inff();
  for (int i = tid; i < SLEN; i += 256) m = fmaxf(m, p[i]);
  red[tid] = m; __syncthreads();
  for (int s = 128; s > 0; s >>= 1) {
    if (tid < s) red[tid] = fmaxf(red[tid], red[tid + s]);
    __syncthreads();
  }
  m = red[0]; __syncthreads();

  float sum = 0.0f;
  for (int i = tid; i < SLEN; i += 256) {
    const float e = __expf(p[i] - m);
    p[i] = e;
    sum += e;
  }
  red[tid] = sum; __syncthreads();
  for (int s = 128; s > 0; s >>= 1) {
    if (tid < s) red[tid] += red[tid + s];
    __syncthreads();
  }
  const float inv = 1.0f / red[0];
  for (int i = tid; i < SLEN; i += 256) p[i] *= inv;
}

// -------------------------- Kernel 5: context ------------------------------
// grid: (SLEN/64, BS*H), block: 128 = 4 waves; V^T tile (64 x 32) shared via
// async-staged LDS; each wave accumulates its own 16x64 context tile.
// Context is written as f16 (consumed only as a WMMA A-operand).
__global__ void k_ctx(const float* __restrict__ wts,
                      const _Float16* __restrict__ vt,
                      _Float16* __restrict__ ctxh) {
  __shared__ alignas(32) _Float16 ldsV[DK * 32];   // 4 KB: (d, s_local)

  const int tid  = threadIdx.x;
  const int lane = tid & 31;
  const int wave = tid >> 5;
  const int qt = blockIdx.x * 4 + wave;
  const int bh = blockIdx.y;
  const int b = bh / H, h = bh % H;

  const float* Wrow  = wts + (size_t)bh * SLEN * SLEN + (size_t)qt * 16 * SLEN;
  const _Float16* VT = vt  + (size_t)bh * DK * SLEN;   // (DK, SLEN)

  // Staging map: 256 chunks of 16B; each thread moves 2.
  const int d0 = tid >> 1;                 // rows 0..63
  const int c0 = (tid & 1) * 8;            // half-index 0 / 8 (of 32)

  v8f acc[4] = {};
  for (int k0 = 0; k0 < SLEN; k0 += 32) {
    __syncthreads();
    stage16B(&ldsV[d0 * 32 + c0],      VT + (size_t)d0 * SLEN + k0 + c0);
    stage16B(&ldsV[d0 * 32 + c0 + 16], VT + (size_t)d0 * SLEN + k0 + c0 + 16);
    stage_wait();
    __syncthreads();

    if (k0 + 32 < SLEN)
      __builtin_prefetch(VT + (size_t)(tid >> 1) * SLEN + k0 + 32 + (tid & 1) * 16, 0, 0);

    v16h a = load_a_f32(Wrow, SLEN, 0, k0, lane);   // weights f32 -> f16
#pragma unroll
    for (int j = 0; j < 4; ++j) {
      v16h bfr = load_bt_lds(ldsV, 32, j * 16, 0, lane);  // B[k][n]=V[s][d]
      acc[j] = wmma16(a, bfr, acc[j]);
    }
  }
  const int half = lane >> 4, n = lane & 15;
#pragma unroll
  for (int j = 0; j < 4; ++j)
#pragma unroll
    for (int r = 0; r < 8; ++r) {
      const int srow = qt * 16 + r + 8 * half;
      const int d = j * 16 + n;
      ctxh[((size_t)b * SLEN + srow) * INNER + h * DK + d] = (_Float16)acc[j][r];
    }
}

// ------------------------- Kernel 6: out projection ------------------------
// grid: (DM/16, BS*SLEN/16), block: 32. out = ctx @ Wo + bo (Wo from wt[3]).
__global__ void k_oproj(const _Float16* __restrict__ ctxh,
                        const _Float16* __restrict__ wt,
                        const float* __restrict__ bo,
                        float* __restrict__ out) {
  const int lane = threadIdx.x;
  const int tn = blockIdx.x, tm = blockIdx.y;
  const int row0 = tm * 16, col0 = tn * 16;
  const _Float16* WoT = wt + (size_t)3 * INNER * DM;   // (DM_out, INNER)

  v8f acc = {};
#pragma unroll 4
  for (int k0 = 0; k0 < INNER; k0 += 32) {
    v16h a = load_a_f16(ctxh, INNER, row0, k0, lane);
    v16h b = load_bt_frag(WoT, INNER, col0, k0, lane);
    acc = wmma16(a, b, acc);
  }
  const int half = lane >> 4, n = lane & 15;
#pragma unroll
  for (int r = 0; r < 8; ++r) {
    const int grow = row0 + r + 8 * half;
    const int gcol = col0 + n;
    out[(size_t)grow * DM + gcol] = acc[r] + bo[gcol];
  }
}

// ------------------------------- launcher ----------------------------------
extern "C" void kernel_launch(void* const* d_in, const int* in_sizes, int n_in,
                              void* d_out, int out_size, void* d_ws, size_t ws_size,
                              hipStream_t stream) {
  const float* x   = (const float*)d_in[0];
  const float* Wq  = (const float*)d_in[1];
  const float* bq  = (const float*)d_in[2];
  const float* Wk  = (const float*)d_in[3];
  const float* bk  = (const float*)d_in[4];
  const float* Wv  = (const float*)d_in[5];
  const float* bv  = (const float*)d_in[6];
  const float* Wo  = (const float*)d_in[7];
  const float* bo  = (const float*)d_in[8];
  const float* rel = (const float*)d_in[9];
  const int*   msk = (const int*)d_in[10];

  // Output tuple: out (b,s,dm) | weights (b,h,q,k) | pos_bias (1,h,q,k).
  float* out = (float*)d_out;
  float* wts = out + (size_t)BS * SLEN * DM;
  float* pb  = wts + (size_t)BS * H * SLEN * SLEN;

  // Workspace: Wt f16 x4 | x f16 | Q,K f16 (b,h,s,d) | V^T f16 | ctx f16.
  const size_t wt_elems  = (size_t)4 * INNER * DM;
  const size_t x_elems   = (size_t)BS * SLEN * DM;
  const size_t qkv_elems = (size_t)BS * H * SLEN * DK;
  _Float16* wt  = (_Float16*)d_ws;
  _Float16* xh  = wt + wt_elems;
  _Float16* qb  = xh + x_elems;
  _Float16* kb  = qb + qkv_elems;
  _Float16* vt  = kb + qkv_elems;
  _Float16* ctx = vt + qkv_elems;

  const dim3 wave(32, 1, 1);

  k_wtrans<<<dim3(DM / 256, INNER, 4), dim3(256), 0, stream>>>(Wq, Wk, Wv, Wo, wt);

  k_xhalf<<<dim3((unsigned)(x_elems / 256)), dim3(256), 0, stream>>>(x, xh);

  k_qkv<<<dim3(INNER / 16, (BS * SLEN) / 16, 3), wave, 0, stream>>>(
      xh, wt, bq, bk, bv, qb, kb, vt);

  const size_t pb_elems = (size_t)H * SLEN * SLEN;
  k_pbias<<<dim3((unsigned)(pb_elems / 256)), dim3(256), 0, stream>>>(rel, pb);

  k_scores<<<dim3(SLEN / 64, BS * H), dim3(128), 0, stream>>>(qb, kb, pb, msk, wts);

  k_softmax<<<dim3(BS * H * SLEN), dim3(256), 0, stream>>>(wts);

  k_ctx<<<dim3(SLEN / 64, BS * H), dim3(128), 0, stream>>>(wts, vt, ctx);

  k_oproj<<<dim3(DM / 16, (BS * SLEN) / 16), wave, 0, stream>>>(ctx, wt, bo, out);
}
